// GRULocalizationClassifier_2473901162627
// MI455X (gfx1250) — matrix operand order repaired and checked by reading, not compiled, not run on hardware
//
#include <hip/hip_runtime.h>
#include <hip/hip_bf16.h>
#include <math.h>

#define SEQ_LEN 32768
#define EMBED   256
#define HIDDEN  512
#define G3      (3 * HIDDEN)      // 1536
#define NWG     16                // workgroups cooperating on the recurrence
#define UNITS   32                // HIDDEN / NWG hidden units owned per WGP
#define ROWS    96                // 3 * UNITS  W_hh rows needed per WGP
#define NTHR    192               // 6 waves of 32; one wave per 16-row WMMA tile

typedef __bf16 v16bf __attribute__((ext_vector_type(16)));
typedef float  v8f   __attribute__((ext_vector_type(8)));
typedef int    v4i   __attribute__((__vector_size__(16)));
typedef v4i __attribute__((address_space(1)))* gv4i_p;   // global int4*
typedef v4i __attribute__((address_space(3)))* lv4i_p;   // LDS int4*

// gfx1250 async global->LDS path (ASYNCcnt), if this toolchain exposes it.
#if defined(__has_builtin)
#if __has_builtin(__builtin_amdgcn_global_load_async_to_lds_b128) && \
    __has_builtin(__builtin_amdgcn_s_wait_asynccnt)
#define USE_ASYNC_LDS 1
#endif
#endif
#ifndef USE_ASYNC_LDS
#define USE_ASYNC_LDS 0
#endif

union BfVec { unsigned int u[8]; v16bf v; };

__device__ __forceinline__ unsigned short f2bf(float f) {
  unsigned int u = __float_as_uint(f);
  unsigned int r = u + 0x7FFFu + ((u >> 16) & 1u);   // round-to-nearest-even
  return (unsigned short)(r >> 16);
}
__device__ __forceinline__ unsigned int pack2bf(float lo, float hi) {
  return (unsigned int)f2bf(lo) | ((unsigned int)f2bf(hi) << 16);
}
__device__ __forceinline__ float sigmoidf_(float x) {
  return 1.0f / (1.0f + __expf(-x));
}

// ---------------------------------------------------------------------------
// Kernel A: gx table.  sequence has only VOCAB=6 distinct values, so the big
// (S x E)@(E x 3H) GEMM collapses to a 6 x 1536 table: emb @ W_ih^T + b_ih.
// ---------------------------------------------------------------------------
__global__ void gru_gx_table(const float* __restrict__ emb,
                             const float* __restrict__ W_ih,
                             const float* __restrict__ b_ih,
                             float* __restrict__ gxTab) {
  int idx = blockIdx.x * blockDim.x + threadIdx.x;
  if (idx >= 6 * G3) return;
  int v = idx / G3, j = idx % G3;
  const float* e = emb  + v * EMBED;
  const float* w = W_ih + j * EMBED;
  float s = b_ih[j];
  for (int k = 0; k < EMBED; ++k) s = fmaf(e[k], w[k], s);
  gxTab[idx] = s;
}

// ---------------------------------------------------------------------------
// Kernel B: persistent distributed GRU recurrence across NWG=16 WGPs.
// WGP w owns hidden units [32w, 32w+32) and the 96 W_hh rows feeding them.
// Each of the 6 waves keeps its 16x512 W_hh slice register-resident as bf16
// WMMA A-fragments (128 VGPRs/lane).  Per step: 16 chained
// v_wmma_f32_16x16x32_bf16 with B = h replicated across all 16 columns
// (every D column is then the matvec), gate math in fp32, h chunk broadcast
// through L2 + flags grid-barrier; h_t pulled back via async global->LDS.
// ---------------------------------------------------------------------------
__global__ void __launch_bounds__(NTHR, 1)
gru_recurrence(const int*   __restrict__ seq,
               const float* __restrict__ W_hh,
               const float* __restrict__ b_hh,
               const float* __restrict__ W_out,
               const float* __restrict__ b_out,
               const float* __restrict__ gxTab,
               unsigned int*   flags,     // [SEQ_LEN], memset 0 per launch
               unsigned short* hbuf,      // [2][HIDDEN] bf16 ping-pong
               float*          hfinal,    // [HIDDEN] fp32
               float*          out)       // [8]
{
  const int tid  = threadIdx.x;
  const int w    = blockIdx.x;        // 0..15
  const int wv   = tid >> 5;          // wave/tile id 0..5
  const int lane = tid & 31;

  __shared__ float ldsGX[6 * ROWS];   // gx table slice (incl. b_ih), per vocab
  __shared__ float ldsBhh[ROWS];      // b_hh slice
  __shared__ float ldsGH[ROWS];       // W_hh @ h results this step
  __shared__ float ldsHown[UNITS];    // fp32 copy of owned hidden units
  __shared__ alignas(64) unsigned short ldsH[HIDDEN];  // bf16 h_{t-1}, full

  // ---- one-time init -----------------------------------------------------
  for (int i = tid; i < 6 * ROWS; i += NTHR) {
    int v = i / ROWS, l = i % ROWS;
    int col = (l / UNITS) * HIDDEN + w * UNITS + (l % UNITS);
    ldsGX[i] = gxTab[v * G3 + col];
  }
  for (int i = tid; i < ROWS; i += NTHR) {
    int col = (i / UNITS) * HIDDEN + w * UNITS + (i % UNITS);
    ldsBhh[i] = b_hh[col];
  }
  for (int i = tid; i < UNITS; i += NTHR) ldsHown[i] = 0.0f;
  {
    unsigned int* p = (unsigned int*)ldsH;
    for (int i = tid; i < HIDDEN / 2; i += NTHR) p[i] = 0u;  // h_{-1} = 0
  }

  // ---- build register-resident A fragments (tile = this wave) ------------
  // A 16x32 layout (contiguous k): lanes 0-15 row M=lane: K=0..15;
  // lanes 16-31 row M=lane-16: K=16..31 (matches B load below).
  v16bf Afrag[16];
  {
    const int grp = lane >> 4;
    const int m   = lane & 15;
    const int l   = wv * 16 + m;                                // local row 0..95
    const int grow = (l / UNITS) * HIDDEN + w * UNITS + (l % UNITS);
    const float* wrow = W_hh + (size_t)grow * HIDDEN;
#pragma unroll
    for (int c = 0; c < 16; ++c) {
      BfVec bv;
      const int k0 = c * 32 + grp * 16;
#pragma unroll
      for (int d = 0; d < 4; ++d)
        bv.u[d]     = pack2bf(wrow[k0 + 2 * d],      wrow[k0 + 2 * d + 1]);
#pragma unroll
      for (int d = 0; d < 4; ++d)
        bv.u[4 + d] = pack2bf(wrow[k0 + 8 + 2 * d], wrow[k0 + 8 + 2 * d + 1]);
      Afrag[c] = bv.v;
    }
  }
  __syncthreads();

  // ---- sequential scan ---------------------------------------------------
  int tok = 0;
  for (int t = 0; t < SEQ_LEN; ++t) {
    // issue token load early; it overlaps the matvec below
    if (tid < UNITS) tok = seq[t];

    // gh_tile(16) = W_tile(16x512) * h(512) via 16 chained WMMA (K=32 each).
    // B has h replicated in all 16 columns -> every D column == matvec.
    {
      v8f acc = {0.f, 0.f, 0.f, 0.f, 0.f, 0.f, 0.f, 0.f};
      const unsigned short* hp = ldsH + ((lane >= 16) ? 16 : 0);
#pragma unroll
      for (int c = 0; c < 16; ++c) {
        v16bf b = *(const v16bf*)(hp + c * 32);   // 32B LDS load, bcast in group
        acc = __builtin_amdgcn_wmma_f32_16x16x32_bf16(
            false, Afrag[c], false, b, (short)0, acc, false, false);
      }
      // D layout: lanes 0-15 VGPR r -> row r; lanes 16-31 VGPR r -> row 8+r
      if (lane == 0) {
#pragma unroll
        for (int r = 0; r < 8; ++r) ldsGH[wv * 16 + r] = acc[r];
      } else if (lane == 16) {
#pragma unroll
        for (int r = 0; r < 8; ++r) ldsGH[wv * 16 + 8 + r] = acc[r];
      }
    }
    __syncthreads();

    // gates + state update for the 32 owned units (wave 0)
    if (tid < UNITS) {
      const int i = tid;
      const float* gx = ldsGX + tok * ROWS;
      float rg = sigmoidf_(gx[i]      + ldsBhh[i]      + ldsGH[i]);
      float zg = sigmoidf_(gx[32 + i] + ldsBhh[32 + i] + ldsGH[32 + i]);
      float ng = tanhf(gx[64 + i] + rg * (ldsGH[64 + i] + ldsBhh[64 + i]));
      float hn = (1.0f - zg) * ng + zg * ldsHown[i];
      ldsHown[i] = hn;
      hbuf[(t & 1) * HIDDEN + w * UNITS + i] = f2bf(hn);
      if (t == SEQ_LEN - 1) hfinal[w * UNITS + i] = hn;
      __threadfence();
      if (i == 0)
        __hip_atomic_fetch_add(&flags[t], 1u, __ATOMIC_RELEASE,
                               __HIP_MEMORY_SCOPE_AGENT);
    }
    // grid barrier: wait for all 16 WGPs' h chunks
    if (tid == 0) {
      while (__hip_atomic_load(&flags[t], __ATOMIC_ACQUIRE,
                               __HIP_MEMORY_SCOPE_AGENT) < NWG)
        __builtin_amdgcn_s_sleep(1);
    }
    __syncthreads();
    // pull full h_t (1KB bf16) into LDS for next step's matvec
#if USE_ASYNC_LDS
    if (tid < 64) {   // waves 0-1: 64 lanes x 16B = 1KB, direct global->LDS
      gv4i_p g = (gv4i_p)((char*)(hbuf + (t & 1) * HIDDEN) + tid * 16);
      lv4i_p l = (lv4i_p)((char*)ldsH + tid * 16);
      __builtin_amdgcn_global_load_async_to_lds_b128(g, l, 0, 0);
      __builtin_amdgcn_s_wait_asynccnt(0);
    }
#else
    {
      const unsigned int* src = (const unsigned int*)(hbuf + (t & 1) * HIDDEN);
      unsigned int* dst = (unsigned int*)ldsH;
      for (int i = tid; i < HIDDEN / 2; i += NTHR) dst[i] = src[i];
    }
#endif
    __syncthreads();
  }

  // ---- head: out = W_out @ hT + b_out (hfinal fully visible after last flag)
  if (w == 0 && tid < 8) {
    float s = b_out[tid];
    const float* wr = W_out + tid * HIDDEN;
    for (int k = 0; k < HIDDEN; ++k) s = fmaf(wr[k], hfinal[k], s);
    out[tid] = s;
  }
}

// ---------------------------------------------------------------------------
// d_ws layout: [gxTab 6*1536 f32 | flags 32768 u32 | hbuf 2*512 bf16 |
//               hfinal 512 f32]  (~172 KB total)
// ---------------------------------------------------------------------------
extern "C" void kernel_launch(void* const* d_in, const int* in_sizes, int n_in,
                              void* d_out, int out_size, void* d_ws,
                              size_t ws_size, hipStream_t stream) {
  const int*   seq   = (const int*)d_in[0];
  const float* emb   = (const float*)d_in[1];
  const float* W_ih  = (const float*)d_in[2];
  const float* W_hh  = (const float*)d_in[3];
  const float* b_ih  = (const float*)d_in[4];
  const float* b_hh  = (const float*)d_in[5];
  const float* W_out = (const float*)d_in[6];
  const float* b_out = (const float*)d_in[7];
  float* out = (float*)d_out;

  char* ws = (char*)d_ws;
  float*          gxTab  = (float*)ws;                            // 36864 B
  unsigned int*   flags  = (unsigned int*)(ws + 36864);           // 131072 B
  unsigned short* hbuf   = (unsigned short*)(ws + 36864 + 131072);// 2048 B
  float*          hfinal = (float*)(ws + 36864 + 131072 + 2048);  // 2048 B

  // barrier flags must start at 0 every launch (graph-capture safe)
  (void)hipMemsetAsync(flags, 0, SEQ_LEN * sizeof(unsigned int), stream);

  gru_gx_table<<<(6 * G3 + 255) / 256, 256, 0, stream>>>(emb, W_ih, b_ih,
                                                         gxTab);
  gru_recurrence<<<NWG, NTHR, 0, stream>>>(seq, W_hh, b_hh, W_out, b_out,
                                           gxTab, flags, hbuf, hfinal, out);
}
